// GraphAttentionLayer_11364483465750
// MI455X (gfx1250) — matrix operand order repaired and checked by reading, not compiled, not run on hardware
//
#include <hip/hip_runtime.h>
#include <hip/hip_bf16.h>

// GAT layer, MI455X (gfx1250), wave32.
// Roofline: adj stream = 134MB -> ~5.8us floor @23.3TB/s; total FLOPs 4.6G ->
// <2us in bf16 WMMA. Design: single fused pass over adj; bf16 WMMA, f32 acc.
// exp(leakyrelu(s1+s2)) factors into exp(s1)exp(s2) / exp(a*s1)exp(a*s2), so
// the hot loop has NO transcendentals (16-bit WMMA shares the TRANS slot per
// ISA 8.4 -- keep that slot for WMMA). hT[batch] (256KB) is staged to LDS via
// the CDNA5 async global->LDS path so the global pipe carries only adj.

typedef __attribute__((ext_vector_type(16))) __bf16 v16bf;
typedef __attribute__((ext_vector_type(8)))  float  v8f;
typedef int v4i __attribute__((vector_size(16)));     // native int4 vector

#define AS1 __attribute__((address_space(1)))         // global
#define AS3 __attribute__((address_space(3)))         // LDS

union V16B { v16bf v; uint4 u[2]; };

#define N_NODES 2048
#define F_IN    128
#define F_OUT   64
#define N_BATCH 8
#define ALPHA   0.2f

// padded LDS row pitch for hT: 2056 bf16 = 4112B -> dword stride 1028 = 64q+4,
// so 16 lanes reading 16 rows land on distinct bank groups for ds_load_b128.
#define HT_STRIDE 2056
#define HT_LDS_BYTES (F_OUT * HT_STRIDE * 2)                  // 263168
#define SMEM_BYTES   (HT_LDS_BYTES + 3 * N_NODES * 4)         // + s2/e2p/e2n

#if defined(__has_builtin)
# if __has_builtin(__builtin_amdgcn_global_load_async_to_lds_b128) && \
     __has_builtin(__builtin_amdgcn_s_wait_asynccnt)
#  define USE_ASYNC_LDS 1
# endif
#endif
#ifndef USE_ASYNC_LDS
# define USE_ASYNC_LDS 0
#endif

static __device__ __forceinline__ v8f wmma_bf16(v16bf a, v16bf b, v8f c) {
    return __builtin_amdgcn_wmma_f32_16x16x32_bf16(
        /*neg_a=*/false, a, /*neg_b=*/false, b,
        /*c_mod=*/(short)0, c, /*reuse_a=*/false, /*reuse_b=*/false);
}

// async 16B global -> LDS copy (per-lane), tracked by ASYNCcnt.
// Builtin signature (from hipcc diagnostic): param0 = AS1 v4i* (global src),
// param1 = LDS dst, then imm offset + cpol.
static __device__ __forceinline__ void async_copy_b128(const void* gp, void* lp) {
#if USE_ASYNC_LDS
    __builtin_amdgcn_global_load_async_to_lds_b128(
        (AS1 v4i*)gp, (AS3 v4i*)lp, 0, 0);
#else
    *(uint4*)lp = *(const uint4*)gp;
#endif
}

// ---------------------------------------------------------------------------
// Kernel 1: hT[b][o][i] = (x @ W)[b][i][o]  (bf16, transposed storage)
// ---------------------------------------------------------------------------
__global__ __launch_bounds__(256)
void gat_h_kernel(const float* __restrict__ x, const float* __restrict__ W,
                  __bf16* __restrict__ hT)
{
    __shared__ __bf16 wt[F_OUT * 136];       // W^T, padded rows (272B pitch)
    const int t = threadIdx.x;
#pragma unroll
    for (int it = 0; it < (F_IN * F_OUT) / 256; ++it) {
        int idx = it * 256 + t;
        int k = idx >> 6, o = idx & 63;      // W is [k][o] row-major
        wt[o * 136 + k] = (__bf16)W[idx];
    }
    __syncthreads();

    const int wave = t >> 5, lane = t & 31;
    const int half = lane >> 4, col = lane & 15;
    const int batch = blockIdx.y;
    const int i0 = blockIdx.x * 128 + wave * 16;
    const int arow = i0 + col;

    const float* xrow = x + (size_t)(batch * N_NODES + arow) * F_IN;

    v8f acc[4];
#pragma unroll
    for (int oc = 0; oc < 4; ++oc)
#pragma unroll
        for (int e = 0; e < 8; ++e) acc[oc][e] = 0.0f;

#pragma unroll
    for (int kc = 0; kc < 4; ++kc) {
        const int k0 = kc * 32;
        float4 q0 = *(const float4*)(xrow + k0 + half * 8);
        float4 q1 = *(const float4*)(xrow + k0 + half * 8 + 4);
        float4 q2 = *(const float4*)(xrow + k0 + 16 + half * 8);
        float4 q3 = *(const float4*)(xrow + k0 + 16 + half * 8 + 4);
        v16bf a;
        a[0]=(__bf16)q0.x;  a[1]=(__bf16)q0.y;  a[2]=(__bf16)q0.z;  a[3]=(__bf16)q0.w;
        a[4]=(__bf16)q1.x;  a[5]=(__bf16)q1.y;  a[6]=(__bf16)q1.z;  a[7]=(__bf16)q1.w;
        a[8]=(__bf16)q2.x;  a[9]=(__bf16)q2.y;  a[10]=(__bf16)q2.z; a[11]=(__bf16)q2.w;
        a[12]=(__bf16)q3.x; a[13]=(__bf16)q3.y; a[14]=(__bf16)q3.z; a[15]=(__bf16)q3.w;
#pragma unroll
        for (int oc = 0; oc < 4; ++oc) {
            const __bf16* bp = &wt[(oc * 16 + col) * 136 + k0 + half * 8];
            V16B bb; bb.u[0] = *(const uint4*)bp; bb.u[1] = *(const uint4*)(bp + 16);
            acc[oc] = wmma_bf16(a, bb.v, acc[oc]);
        }
    }

#pragma unroll
    for (int oc = 0; oc < 4; ++oc) {
        const int o = oc * 16 + col;
        __bf16* hp = hT + ((size_t)batch * F_OUT + o) * N_NODES + i0 + half * 8;
#pragma unroll
        for (int r = 0; r < 8; ++r) hp[r] = (__bf16)acc[oc][r];
    }
}

// ---------------------------------------------------------------------------
// Kernel 2: s1/s2 dots plus their exp factorizations:
//   e1p=exp(s1) e1n=exp(a*s1) e2p=exp(s2) e2n=exp(a*s2)
// so the attention hot loop needs zero transcendentals.
// ---------------------------------------------------------------------------
__global__ __launch_bounds__(256)
void gat_s_kernel(const __bf16* __restrict__ hT, const float* __restrict__ a,
                  float* __restrict__ s1, float* __restrict__ e1p, float* __restrict__ e1n,
                  float* __restrict__ s2, float* __restrict__ e2p, float* __restrict__ e2n)
{
    const int t = blockIdx.x * 256 + threadIdx.x;          // 16384 threads
    const int b = t >> 11, i = t & (N_NODES - 1);
    const __bf16* hp = hT + (size_t)b * F_OUT * N_NODES + i;
    float r1 = 0.0f, r2 = 0.0f;
#pragma unroll 8
    for (int o = 0; o < F_OUT; ++o) {
        float hv = (float)hp[(size_t)o * N_NODES];
        r1 = fmaf(hv, a[o], r1);
        r2 = fmaf(hv, a[F_OUT + o], r2);
    }
    s1[t] = r1; e1p[t] = __expf(r1); e1n[t] = __expf(ALPHA * r1);
    s2[t] = r2; e2p[t] = __expf(r2); e2n[t] = __expf(ALPHA * r2);
}

// ---------------------------------------------------------------------------
// Kernel 3: fused mask -> leaky_relu -> exp -> normalized attn @ h.
// One adj pass. 5 WMMAs per 32-wide K chunk (4 output chunks + ones-B giving
// the softmax denominator in the *same* C layout -> lane-local divide).
// hT[batch] staged to LDS with async global->LDS (ASYNCcnt).
// ---------------------------------------------------------------------------
static __device__ __forceinline__ __bf16 edge_w(int av, float s2v, float p2, float n2,
                                                float s1r, float p1, float n1) {
    float t  = s1r + s2v;
    float e2 = (t > 0.0f) ? p2 : n2;       // exp(s2) vs exp(a*s2)
    float e1 = (t > 0.0f) ? p1 : n1;       // exp(s1) vs exp(a*s1)
    float w  = (av > 0) ? e1 * e2 : 0.0f;  // mask == weight 0
    return (__bf16)w;
}

__global__ __launch_bounds__(256)
void gat_attn_kernel(const int* __restrict__ adj, const __bf16* __restrict__ hT,
                     const float* __restrict__ s1, const float* __restrict__ e1p,
                     const float* __restrict__ e1n, const float* __restrict__ s2,
                     const float* __restrict__ e2p, const float* __restrict__ e2n,
                     float* __restrict__ out)
{
    extern __shared__ __align__(16) char smem[];
    __bf16* hT_l   = (__bf16*)smem;                         // 64 x 2056 (padded)
    float*  s2l    = (float*)(smem + HT_LDS_BYTES);         // 2048
    float*  e2pl   = s2l + N_NODES;
    float*  e2nl   = e2pl + N_NODES;

    const int t = threadIdx.x;
    const int batch = blockIdx.y;

    // ---- stage hT[batch] (256KB) to LDS: async global->LDS B128 per lane ----
    {
        const __bf16* hsrc = hT + (size_t)batch * F_OUT * N_NODES;
#pragma unroll 4
        for (int it = 0; it < (F_OUT * N_NODES) / (256 * 8); ++it) {  // 64 iters
            int e = (it * 256 + t) * 8;                    // 8 bf16 = 16B
            int o = e >> 11, j = e & (N_NODES - 1);
            const __bf16* gp = hsrc + ((size_t)o << 11) + j;
            __bf16* lp = hT_l + o * HT_STRIDE + j;
            async_copy_b128(gp, lp);
        }
        // s2-side arrays (24KB): plain copies
#pragma unroll
        for (int it = 0; it < N_NODES / 256; ++it) {
            int idx = it * 256 + t;
            s2l[idx]  = s2[batch * N_NODES + idx];
            e2pl[idx] = e2p[batch * N_NODES + idx];
            e2nl[idx] = e2n[batch * N_NODES + idx];
        }
#if USE_ASYNC_LDS
        __builtin_amdgcn_s_wait_asynccnt(0);
#endif
        __syncthreads();
    }

    const int wave = t >> 5, lane = t & 31;
    const int half = lane >> 4, col = lane & 15;
    const int i0 = (blockIdx.x * 8 + wave) * 16;
    const int arowi = i0 + col;                             // A-operand row
    const float s1r = s1[batch * N_NODES + arowi];
    const float p1  = e1p[batch * N_NODES + arowi];
    const float n1  = e1n[batch * N_NODES + arowi];

    const int* arow = adj + ((size_t)batch * N_NODES + arowi) * N_NODES;

    v16bf ones = {(__bf16)1.0f,(__bf16)1.0f,(__bf16)1.0f,(__bf16)1.0f,
                  (__bf16)1.0f,(__bf16)1.0f,(__bf16)1.0f,(__bf16)1.0f,
                  (__bf16)1.0f,(__bf16)1.0f,(__bf16)1.0f,(__bf16)1.0f,
                  (__bf16)1.0f,(__bf16)1.0f,(__bf16)1.0f,(__bf16)1.0f};

    v8f acc[4], rs;
#pragma unroll
    for (int e = 0; e < 8; ++e) rs[e] = 0.0f;
#pragma unroll
    for (int oc = 0; oc < 4; ++oc)
#pragma unroll
        for (int e = 0; e < 8; ++e) acc[oc][e] = 0.0f;

    for (int j0 = 0; j0 < N_NODES; j0 += 32) {
        const int* ap = arow + j0 + half * 8;
        int4 q0 = *(const int4*)ap;
        int4 q1 = *(const int4*)(ap + 4);
        int4 q2 = *(const int4*)(ap + 16);
        int4 q3 = *(const int4*)(ap + 20);
        __builtin_prefetch(ap + 128, 0, 1);                 // adj stream prefetch

        const float* sp = s2l  + j0 + half * 8;
        const float* pp = e2pl + j0 + half * 8;
        const float* np = e2nl + j0 + half * 8;
        float4 f0 = *(const float4*)sp,        f1 = *(const float4*)(sp + 4);
        float4 f2 = *(const float4*)(sp + 16), f3 = *(const float4*)(sp + 20);
        float4 p0 = *(const float4*)pp,        pq1 = *(const float4*)(pp + 4);
        float4 p2 = *(const float4*)(pp + 16), p3 = *(const float4*)(pp + 20);
        float4 n0 = *(const float4*)np,        nq1 = *(const float4*)(np + 4);
        float4 n2 = *(const float4*)(np + 16), n3 = *(const float4*)(np + 20);

        v16bf a;
        a[0]  = edge_w(q0.x, f0.x, p0.x,  n0.x,  s1r, p1, n1);
        a[1]  = edge_w(q0.y, f0.y, p0.y,  n0.y,  s1r, p1, n1);
        a[2]  = edge_w(q0.z, f0.z, p0.z,  n0.z,  s1r, p1, n1);
        a[3]  = edge_w(q0.w, f0.w, p0.w,  n0.w,  s1r, p1, n1);
        a[4]  = edge_w(q1.x, f1.x, pq1.x, nq1.x, s1r, p1, n1);
        a[5]  = edge_w(q1.y, f1.y, pq1.y, nq1.y, s1r, p1, n1);
        a[6]  = edge_w(q1.z, f1.z, pq1.z, nq1.z, s1r, p1, n1);
        a[7]  = edge_w(q1.w, f1.w, pq1.w, nq1.w, s1r, p1, n1);
        a[8]  = edge_w(q2.x, f2.x, p2.x,  n2.x,  s1r, p1, n1);
        a[9]  = edge_w(q2.y, f2.y, p2.y,  n2.y,  s1r, p1, n1);
        a[10] = edge_w(q2.z, f2.z, p2.z,  n2.z,  s1r, p1, n1);
        a[11] = edge_w(q2.w, f2.w, p2.w,  n2.w,  s1r, p1, n1);
        a[12] = edge_w(q3.x, f3.x, p3.x,  n3.x,  s1r, p1, n1);
        a[13] = edge_w(q3.y, f3.y, p3.y,  n3.y,  s1r, p1, n1);
        a[14] = edge_w(q3.z, f3.z, p3.z,  n3.z,  s1r, p1, n1);
        a[15] = edge_w(q3.w, f3.w, p3.w,  n3.w,  s1r, p1, n1);

        rs = wmma_bf16(a, ones, rs);                        // denominator
#pragma unroll
        for (int oc = 0; oc < 4; ++oc) {
            const __bf16* bp = hT_l + (oc * 16 + col) * HT_STRIDE + j0 + half * 8;
            V16B bb; bb.u[0] = *(const uint4*)bp; bb.u[1] = *(const uint4*)(bp + 16);
            acc[oc] = wmma_bf16(a, bb.v, acc[oc]);
        }
    }

    float inv[8];
#pragma unroll
    for (int r = 0; r < 8; ++r) inv[r] = (rs[r] > 0.0f) ? (1.0f / rs[r]) : 0.0f;

#pragma unroll
    for (int oc = 0; oc < 4; ++oc) {
        float* op = out + ((size_t)batch * N_NODES + i0 + half * 8) * F_OUT + oc * 16 + col;
#pragma unroll
        for (int r = 0; r < 8; ++r)
            op[(size_t)r * F_OUT] = acc[oc][r] * inv[r];
    }
}

// ---------------------------------------------------------------------------
extern "C" void kernel_launch(void* const* d_in, const int* in_sizes, int n_in,
                              void* d_out, int out_size, void* d_ws, size_t ws_size,
                              hipStream_t stream)
{
    (void)in_sizes; (void)n_in; (void)out_size; (void)ws_size;
    const float* x   = (const float*)d_in[0];   // (8,2048,128) f32
    const int*   adj = (const int*)d_in[1];     // (8,2048,2048) i32
    const float* W   = (const float*)d_in[2];   // (128,64) f32
    const float* a   = (const float*)d_in[3];   // (128,1) f32
    float* out = (float*)d_out;                 // (8,2048,64) f32

    char* ws = (char*)d_ws;
    __bf16* hT = (__bf16*)ws;                                       // 2 MB
    float* s1  = (float*)(ws + (size_t)N_BATCH * F_OUT * N_NODES * 2);
    float* e1p = s1  + N_BATCH * N_NODES;
    float* e1n = e1p + N_BATCH * N_NODES;
    float* s2  = e1n + N_BATCH * N_NODES;
    float* e2p = s2  + N_BATCH * N_NODES;
    float* e2n = e2p + N_BATCH * N_NODES;

    // Allow 288KB dynamic LDS (WGP has 320KB). Attribute set is idempotent
    // and capture-safe (no allocation / no stream op).
    (void)hipFuncSetAttribute((const void*)gat_attn_kernel,
                              hipFuncAttributeMaxDynamicSharedMemorySize,
                              (int)SMEM_BYTES);

    gat_h_kernel   <<<dim3(16, N_BATCH), 256, 0, stream>>>(x, W, hT);
    gat_s_kernel   <<<dim3(64),          256, 0, stream>>>(hT, a, s1, e1p, e1n, s2, e2p, e2n);
    gat_attn_kernel<<<dim3(16, N_BATCH), 256, SMEM_BYTES, stream>>>(
        adj, hT, s1, e1p, e1n, s2, e2p, e2n, out);
}
